// CausalSelfAttention_8065948581923
// MI455X (gfx1250) — matrix operand order repaired and checked by reading, not compiled, not run on hardware
//
#include <hip/hip_runtime.h>
#include <hip/hip_bf16.h>
#include <math.h>

// ---------------------------------------------------------------------------
// CausalSelfAttention for MI455X (gfx1250, wave32, WMMA 16x16x32 f16)
// B=4, T=2048, C=1024, H=16, D=64  -> M = B*T = 8192
// Pipeline: fused QKV GEMM (f16 WMMA, f32 acc) -> flash attention -> out proj.
// ---------------------------------------------------------------------------

typedef __attribute__((ext_vector_type(16))) _Float16 v16h;
typedef __attribute__((ext_vector_type(8)))  _Float16 v8h;
typedef __attribute__((ext_vector_type(2)))  _Float16 v2h;
typedef __attribute__((ext_vector_type(8)))  float    v8f;

#define TT 2048
#define HH 16
#define DD 64
#define CC 1024
#define BB 4
#define MM (BB * TT)

// LDS byte-offset of a generic pointer that points into shared memory
// (generic -> address_space(3) cast; LDS pointers are 32-bit offsets).
__device__ __forceinline__ unsigned lds_addr(const void* p) {
    return (unsigned)(size_t)(__attribute__((address_space(3))) const void*)p;
}

// Async DMA: LDS[dst + imm] = MEM[src + imm], 16B per lane, ASYNCcnt-tracked.
#define ASYNC_B128(dst, src, imm)                                              \
    asm volatile("global_load_async_to_lds_b128 %0, %1, off offset:" #imm      \
                 :: "v"(dst), "v"(src) : "memory")
#define WAIT_ASYNC() asm volatile("s_wait_asynccnt 0x0" ::: "memory")

// ---- WMMA operand loaders --------------------------------------------------
// A operand (16x32 f16, M x K): lane holds row M = lane&15; halves i<8 ->
// K = klo+i, i>=8 -> K = klo+16+(i-8); klo = +8 for lanes 16..31.
__device__ __forceinline__ v16h load_a16(const _Float16* rowp, int k0, int lane) {
    int klo = k0 + ((lane & 16) ? 8 : 0);
    v8h lo = *(const v8h*)(rowp + klo);
    v8h hi = *(const v8h*)(rowp + klo + 16);
    v16h a;
#pragma unroll
    for (int i = 0; i < 8; ++i) { a[i] = lo[i]; a[i + 8] = hi[i]; }
    return a;
}

// B operand (32x16 f16, K x N) from an LDS tile stored [K][N] row-major:
// lane holds N = n0 + (lane&15); halves i<8 -> K = klo+i, i>=8 -> K = klo+16+(i-8);
// klo = k0 (+8 for lanes 16..31).
__device__ __forceinline__ v16h load_b16(const _Float16* base, int ldb, int k0,
                                         int n0, int lane) {
    const int klo = k0 + ((lane & 16) ? 8 : 0);
    const _Float16* p = base + klo * ldb + n0 + (lane & 15);
    v16h b;
#pragma unroll
    for (int i = 0; i < 8; ++i) { b[i] = p[i * ldb]; b[i + 8] = p[(i + 16) * ldb]; }
    return b;
}

__device__ __forceinline__ v8f wmma32(v16h a, v16h b, v8f c) {
    return __builtin_amdgcn_wmma_f32_16x16x32_f16(false, a, false, b,
                                                  (short)0, c, false, false);
}

// ---------------------------------------------------------------------------
// Fused QKV projection: q/k/v = x @ W{q,k,v}^T + b, scattered f16 to [B,H,T,D].
// Block: 128 threads (4 waves), 64x64 tile, K-step 32; one shared A tile
// feeds 12 WMMAs (3 outputs x 4 N-tiles) per K-step.
// ---------------------------------------------------------------------------
__global__ __launch_bounds__(128) void gemm_qkv_fused(
    const float* __restrict__ X,
    const float* __restrict__ Wq, const float* __restrict__ Wk,
    const float* __restrict__ Wv,
    const float* __restrict__ bq, const float* __restrict__ bk,
    const float* __restrict__ bv,
    _Float16* __restrict__ qh, _Float16* __restrict__ kh,
    _Float16* __restrict__ vh) {
    __shared__ _Float16 As[64 * 32];         // [row][k]
    __shared__ _Float16 Bs[3][32 * 72];      // [k][n], stride 72 (16B rows)

    const int tid  = threadIdx.x;
    const int lane = tid & 31;
    const int wv   = tid >> 5;
    const int m0   = blockIdx.x * 64;
    const int n0   = blockIdx.y * 64;

    const float* Ws[3]  = {Wq, Wk, Wv};
    const float* bsp[3] = {bq, bk, bv};
    _Float16*    Os[3]  = {qh, kh, vh};

    v8f acc[3][4] = {};

    const int row = tid >> 1;                // 0..63
    const int kh2 = (tid & 1) * 16;          // 0 or 16

    for (int k0 = 0; k0 < CC; k0 += 32) {
        // ---- stage A tile (64x32) f32 -> f16, packed b32 stores ------------
        {
            const float* src = X + (size_t)(m0 + row) * CC + k0 + kh2;
#pragma unroll
            for (int i = 0; i < 8; ++i) {
                v2h p; p[0] = (_Float16)src[2 * i]; p[1] = (_Float16)src[2 * i + 1];
                *(v2h*)&As[row * 32 + kh2 + 2 * i] = p;
            }
        }
        // ---- stage 3 B tiles transposed: Bs[w][k][n] = W[w][n0+n][k0+k] ----
#pragma unroll
        for (int w = 0; w < 3; ++w) {
            const float* src = Ws[w] + (size_t)(n0 + row) * CC + k0 + kh2;
#pragma unroll
            for (int i = 0; i < 16; ++i)
                Bs[w][(kh2 + i) * 72 + row] = (_Float16)src[i];
        }
        __syncthreads();

        // ---- compute: each wave owns a 16x64 strip, A reused 12x -----------
        const _Float16* arow = &As[(wv * 16 + (lane & 15)) * 32];
        const v16h a = load_a16(arow, 0, lane);
#pragma unroll
        for (int w = 0; w < 3; ++w)
#pragma unroll
            for (int j = 0; j < 4; ++j) {
                v16h b = load_b16(Bs[w], 72, 0, j * 16, lane);
                acc[w][j] = wmma32(a, b, acc[w][j]);
            }
        __syncthreads();
    }

    // ---- epilogue: bias + f16 scatter into [B,H,T,D] -----------------------
    const int grp = (lane & 16) ? 8 : 0;
#pragma unroll
    for (int w = 0; w < 3; ++w)
#pragma unroll
        for (int j = 0; j < 4; ++j) {
            const int col = n0 + j * 16 + (lane & 15);
            const float bv2 = bsp[w][col];
            const int h = col >> 6, d = col & (DD - 1);
#pragma unroll
            for (int r = 0; r < 8; ++r) {
                const int rowg = m0 + wv * 16 + r + grp;
                const int b = rowg >> 11, t = rowg & (TT - 1);
                Os[w][(((size_t)(b * HH + h)) * TT + t) * DD + d] =
                    (_Float16)(acc[w][j][r] + bv2);
            }
        }
}

// ---------------------------------------------------------------------------
// Flash attention: one 4-wave block per (b,h, 64-row q block).  Q strips in
// registers as WMMA A-operands; K^T staged (transposed) in LDS; V staged via
// GLOBAL_LOAD_ASYNC_TO_LDS_B128; online softmax; P re-shaped via LDS bounce.
// ---------------------------------------------------------------------------
__global__ __launch_bounds__(128) void attn_wmma(
    const _Float16* __restrict__ Qh, const _Float16* __restrict__ Kh,
    const _Float16* __restrict__ Vh, _Float16* __restrict__ Yh) {
    __shared__ _Float16 Kt[64 * 72];          // [d][n]
    __shared__ _Float16 Vs[64 * 72];          // [n][d]
    __shared__ _Float16 Ps[4][16 * 72];       // per-wave P strip [row][col]

    const int tid  = threadIdx.x;
    const int lane = tid & 31;
    const int wv   = tid >> 5;
    const int qblk = blockIdx.x;              // 0..31
    const int bh   = blockIdx.y;              // 0..63 (b*H + h)
    const int q0   = qblk * 64;
    const float NEG_INF = -__builtin_huge_valf();

    const _Float16* qrow =
        Qh + ((size_t)bh * TT + q0 + wv * 16 + (lane & 15)) * DD;
    const v16h aq0 = load_a16(qrow, 0, lane);
    const v16h aq1 = load_a16(qrow, 32, lane);

    v8f o[4] = {};
    float mrow[8], lrow[8];
#pragma unroll
    for (int r = 0; r < 8; ++r) { mrow[r] = NEG_INF; lrow[r] = 0.0f; }

    const int grp    = (lane & 16) ? 8 : 0;
    const int rowg0  = q0 + wv * 16 + grp;
    const int ncopy  = tid >> 1;              // 0..63
    const int halfsz = (tid & 1) * 32;        // 0 or 32

    const unsigned vdst = lds_addr(&Vs[ncopy * 72 + halfsz]);

    for (int jb = 0; jb <= qblk; ++jb) {
        const int j0 = jb * 64;
        // ---- stage V asynchronously: Vs[n][d] = V[j0+n][d] -----------------
        {
            const _Float16* src = Vh + ((size_t)bh * TT + j0 + ncopy) * DD + halfsz;
            ASYNC_B128(vdst, src, 0);
            ASYNC_B128(vdst, src, 16);
            ASYNC_B128(vdst, src, 32);
            ASYNC_B128(vdst, src, 48);
        }
        // ---- stage K^T (transpose): Kt[d][n] = K[j0+n][d] ------------------
        {
            const _Float16* src = Kh + ((size_t)bh * TT + j0 + ncopy) * DD + halfsz;
#pragma unroll
            for (int i = 0; i < 32; ++i)
                Kt[(halfsz + i) * 72 + ncopy] = src[i];
        }
        WAIT_ASYNC();
        __syncthreads();

        // ---- S = (Q K^T) * scale, causal mask ------------------------------
        v8f s[4];
#pragma unroll
        for (int jt = 0; jt < 4; ++jt) {
            v8f z = {};
            z = wmma32(aq0, load_b16(Kt, 72, 0,  jt * 16, lane), z);
            s[jt] = wmma32(aq1, load_b16(Kt, 72, 32, jt * 16, lane), z);
        }
        const bool diag = (jb == qblk);
#pragma unroll
        for (int jt = 0; jt < 4; ++jt) {
            const int col = j0 + jt * 16 + (lane & 15);
#pragma unroll
            for (int r = 0; r < 8; ++r) {
                float v = s[jt][r] * 0.125f;              // 1/sqrt(64)
                if (diag && col > (rowg0 + r)) v = NEG_INF;
                s[jt][r] = v;
            }
        }

        // ---- online softmax (rows live across 16-lane groups) --------------
        float alpha[8];
#pragma unroll
        for (int r = 0; r < 8; ++r) {
            float mx = fmaxf(fmaxf(s[0][r], s[1][r]), fmaxf(s[2][r], s[3][r]));
#pragma unroll
            for (int m = 1; m < 16; m <<= 1)
                mx = fmaxf(mx, __shfl_xor(mx, m, 16));
            const float mnew = fmaxf(mrow[r], mx);
            alpha[r] = __expf(mrow[r] - mnew);            // 0 on first pass
            mrow[r]  = mnew;
            float rs = 0.0f;
#pragma unroll
            for (int jt = 0; jt < 4; ++jt) {
                const float p = __expf(s[jt][r] - mnew);
                s[jt][r] = p;
                rs += p;
            }
#pragma unroll
            for (int m = 1; m < 16; m <<= 1)
                rs += __shfl_xor(rs, m, 16);
            lrow[r] = lrow[r] * alpha[r] + rs;
        }
#pragma unroll
        for (int dt = 0; dt < 4; ++dt)
#pragma unroll
            for (int r = 0; r < 8; ++r)
                o[dt][r] *= alpha[r];

        // ---- P: C-layout regs -> per-wave LDS -> A-layout regs -------------
#pragma unroll
        for (int jt = 0; jt < 4; ++jt)
#pragma unroll
            for (int r = 0; r < 8; ++r)
                Ps[wv][(r + grp) * 72 + jt * 16 + (lane & 15)] = (_Float16)s[jt][r];
        asm volatile("s_wait_dscnt 0x0" ::: "memory");
        const _Float16* prow = &Ps[wv][(lane & 15) * 72];
        const v16h ap0 = load_a16(prow, 0, lane);
        const v16h ap1 = load_a16(prow, 32, lane);

        // ---- O += P @ V ----------------------------------------------------
#pragma unroll
        for (int dt = 0; dt < 4; ++dt) {
            o[dt] = wmma32(ap0, load_b16(Vs, 72, 0,  dt * 16, lane), o[dt]);
            o[dt] = wmma32(ap1, load_b16(Vs, 72, 32, dt * 16, lane), o[dt]);
        }
        __syncthreads();
    }

    // ---- normalize + store y in [B,T,C] f16 --------------------------------
    const int b = bh >> 4, h = bh & 15;
#pragma unroll
    for (int r = 0; r < 8; ++r) {
        const float inv = 1.0f / lrow[r];
        const int rowg = rowg0 + r;
#pragma unroll
        for (int dt = 0; dt < 4; ++dt) {
            const int d = dt * 16 + (lane & 15);
            Yh[((size_t)b * TT + rowg) * CC + h * DD + d] =
                (_Float16)(o[dt][r] * inv);
        }
    }
}

// ---------------------------------------------------------------------------
// Output projection: out = y @ Wp^T + bp, y f16 [M,C] (async-staged to LDS),
// output f32 [M,C] to d_out.
// ---------------------------------------------------------------------------
__global__ __launch_bounds__(128) void gemm_proj(
    const _Float16* __restrict__ Yh, const float* __restrict__ W,
    const float* __restrict__ bias, float* __restrict__ out) {
    __shared__ _Float16 As[64 * 32];
    __shared__ _Float16 Bs[32 * 72];

    const int tid  = threadIdx.x;
    const int lane = tid & 31;
    const int wv   = tid >> 5;
    const int m0   = blockIdx.x * 64;
    const int n0   = blockIdx.y * 64;

    v8f acc[4] = {};

    const int row = tid >> 1;
    const int kh2 = (tid & 1) * 16;
    const unsigned adst = lds_addr(&As[row * 32 + kh2]);

    for (int k0 = 0; k0 < CC; k0 += 32) {
        // ---- stage A tile asynchronously (f16 straight copy) ---------------
        {
            const _Float16* src = Yh + (size_t)(m0 + row) * CC + k0 + kh2;
            ASYNC_B128(adst, src, 0);
            ASYNC_B128(adst, src, 16);
        }
        // ---- stage B tile transposed: Bs[k][n] = W[n0+n][k0+k] -------------
        {
            const float* src = W + (size_t)(n0 + row) * CC + k0 + kh2;
#pragma unroll
            for (int i = 0; i < 16; ++i)
                Bs[(kh2 + i) * 72 + row] = (_Float16)src[i];
        }
        WAIT_ASYNC();
        __syncthreads();

        const _Float16* arow = &As[(wv * 16 + (lane & 15)) * 32];
        const v16h a = load_a16(arow, 0, lane);
#pragma unroll
        for (int j = 0; j < 4; ++j) {
            v16h b = load_b16(Bs, 72, 0, j * 16, lane);
            acc[j] = wmma32(a, b, acc[j]);
        }
        __syncthreads();
    }

    const int grp = (lane & 16) ? 8 : 0;
#pragma unroll
    for (int j = 0; j < 4; ++j) {
        const int col = n0 + j * 16 + (lane & 15);
        const float bv = bias[col];
#pragma unroll
        for (int r = 0; r < 8; ++r) {
            const int rowg = m0 + wv * 16 + r + grp;
            out[(size_t)rowg * CC + col] = acc[j][r] + bv;
        }
    }
}

// ---------------------------------------------------------------------------
extern "C" void kernel_launch(void* const* d_in, const int* in_sizes, int n_in,
                              void* d_out, int out_size, void* d_ws, size_t ws_size,
                              hipStream_t stream) {
    (void)in_sizes; (void)n_in; (void)out_size; (void)ws_size;
    const float* x  = (const float*)d_in[0];
    const float* Wq = (const float*)d_in[1];
    const float* bq = (const float*)d_in[2];
    const float* Wk = (const float*)d_in[3];
    const float* bk = (const float*)d_in[4];
    const float* Wv = (const float*)d_in[5];
    const float* bv = (const float*)d_in[6];
    const float* Wp = (const float*)d_in[7];
    const float* bp = (const float*)d_in[8];

    const size_t NEL = (size_t)MM * CC;       // 8,388,608 elements
    _Float16* qh = (_Float16*)d_ws;
    _Float16* kh = qh + NEL;
    _Float16* vh = kh + NEL;
    _Float16* yh = vh + NEL;

    dim3 gg(MM / 64, CC / 64);                // (128, 16)
    gemm_qkv_fused<<<gg, 128, 0, stream>>>(x, Wq, Wk, Wv, bq, bk, bv,
                                           qh, kh, vh);
    attn_wmma<<<dim3(TT / 64, BB * HH), 128, 0, stream>>>(qh, kh, vh, yh);
    gemm_proj<<<gg, 128, 0, stream>>>(yh, Wp, bp, (float*)d_out);
}
